// MambaLayer_12438225289949
// MI455X (gfx1250) — compile-verified, matrix operand order checked by reading, not run on hardware
//
#include <hip/hip_runtime.h>
#include <cstdint>
#include <cstddef>

// ---------------- problem constants ----------------
#define D_MODEL 768
#define D_STATE 16
#define D_CONV  4
#define D_INNER 1536        // 2 * 768
#define DT_RANK 48
#define DTK     64          // dt_rank padded to WMMA-friendly K
#define BATCH   8
#define SEQLEN  2048
#define MROWS   (BATCH * SEQLEN)   // 16384
#define XDBL_W  80          // DT_RANK + 2*D_STATE
#define LDS_STRIDE 48       // 32 + 16 halfs pad (96B rows, 16B aligned)

typedef __bf16 bf16;
typedef __bf16 v16bf __attribute__((ext_vector_type(16)));
typedef __bf16 v8bf  __attribute__((ext_vector_type(8)));
typedef float  v8f   __attribute__((ext_vector_type(8)));

// ---------------- CDNA5 async global -> LDS copy (ASYNCcnt-tracked) ----------------
__device__ __forceinline__ void async_ld_b128(unsigned lds_byte_off, const void* gaddr) {
    // GLOBAL_LOAD_ASYNC_TO_LDS_B128 vdst(lds addr), vaddr(64b global), off (GV mode)
    asm volatile("global_load_async_to_lds_b128 %0, %1, off"
                 :: "v"(lds_byte_off), "v"(gaddr)
                 : "memory");
}
#define WAIT_ASYNC_LE4() asm volatile("s_wait_asynccnt 0x4" ::: "memory")
#define WAIT_ASYNC_0()   asm volatile("s_wait_asynccnt 0x0" ::: "memory")

// ---------------- small prep kernels ----------------
__global__ __launch_bounds__(256) void cvt_f32_bf16_kernel(const float* __restrict__ src,
                                                           bf16* __restrict__ dst, int n) {
    int i = blockIdx.x * 256 + threadIdx.x;
    if (i < n) dst[i] = (bf16)src[i];
}

__global__ __launch_bounds__(256) void pad_dtw_kernel(const float* __restrict__ src,
                                                      bf16* __restrict__ dst) {
    int i = blockIdx.x * 256 + threadIdx.x;           // over D_INNER * DTK
    if (i >= D_INNER * DTK) return;
    int r = i / DTK, c = i % DTK;
    float v = (c < DT_RANK) ? src[r * DT_RANK + c] : 0.f;
    dst[i] = (bf16)v;
}

__global__ __launch_bounds__(256) void prep_A_kernel(const float* __restrict__ alog,
                                                     float* __restrict__ aneg, int n) {
    int i = blockIdx.x * 256 + threadIdx.x;
    if (i < n) aneg[i] = -__expf(alog[i]);
}

__global__ __launch_bounds__(256) void zero_bf16_kernel(bf16* __restrict__ dst, int n) {
    int i = blockIdx.x * 256 + threadIdx.x;
    if (i < n) dst[i] = (bf16)0.f;
}

// ---------------- WMMA fragment load (LDS, row-major, stride 48 halfs) ----------------
// 16-bit A-matrix 16x32 per-lane layout (ISA 7.12.2): lane r (0..15) holds row r,
// K = 0..7 (regs 0..3) and 16..23 (regs 4..7); lanes 16..31 hold K = 8..15 / 24..31.
// B fragment mirrors this against the N x K (transposed-weight) tile, so both are
// contiguous 16B chunks -> ds_load_b128 pairs.
__device__ __forceinline__ v16bf frag_ld(const bf16* tile) {
    const int lane = threadIdx.x & 31;
    const int r = lane & 15;
    const int h = lane >> 4;
    const bf16* p = tile + r * LDS_STRIDE + h * 8;
    v8bf lo = *(const v8bf*)(p);
    v8bf hi = *(const v8bf*)(p + 16);
    v16bf f;
#pragma unroll
    for (int i = 0; i < 8; ++i) { f[i] = lo[i]; f[8 + i] = hi[i]; }
    return f;
}

// ---------------- fused WMMA GEMM: out = A(MxK) * Bt(NxK)^T ----------------
// MODE 0: in_proj  -> split cols into x (out0) / z (out1)
// MODE 1: x_proj   -> out0 = x_dbl (ld 80), cols<48 also to bf16 dt_in (ld 64)
// MODE 2: dt_proj  -> out0 = softplus(acc + bias[col])
// MODE 3: out_proj -> out0 = acc + resid (pre-LN residual)
// Requires: M % 128 == 0, K % 64 == 0. N handled with clamped B rows + store guard.
template <int MODE>
__global__ __launch_bounds__(256, 2) void gemm_bf16_wmma(
    const bf16* __restrict__ Ag, const bf16* __restrict__ Btg, int N, int K,
    float* __restrict__ out0, float* __restrict__ out1,
    const float* __restrict__ bias, const float* __restrict__ resid,
    bf16* __restrict__ outbf)
{
    __shared__ __align__(16) bf16 As[2][128 * LDS_STRIDE];
    __shared__ __align__(16) bf16 Bs[2][128 * LDS_STRIDE];

    const int tid  = threadIdx.x;
    const int wave = tid >> 5;
    const int wm   = wave >> 2;   // 0..1  -> 64-row half
    const int wn   = wave & 3;    // 0..3  -> 32-col slice
    const int m0   = blockIdx.y * 128;
    const int n0   = blockIdx.x * 128;

    v8f acc[4][2];
#pragma unroll
    for (int i = 0; i < 4; ++i)
#pragma unroll
        for (int j = 0; j < 2; ++j)
            acc[i][j] = v8f{0.f, 0.f, 0.f, 0.f, 0.f, 0.f, 0.f, 0.f};

    // per-thread staging slots: 2 x 16B chunks of a 128x32 tile per buffer
    const bf16* pA[2];
    const bf16* pB[2];
    unsigned la[2][2], lb[2][2];          // [buffer][chunk]
#pragma unroll
    for (int i = 0; i < 2; ++i) {
        const int li  = tid + i * 256;
        const int row = li >> 2;          // 0..127
        const int c   = (li & 3) * 8;     // 0,8,16,24 halfs
        pA[i] = Ag + (size_t)(m0 + row) * K + c;
        int gr = n0 + row;
        if (gr >= N) gr = N - 1;          // clamp: cols >= N are never stored
        pB[i] = Btg + (size_t)gr * K + c;
#pragma unroll
        for (int s = 0; s < 2; ++s) {
            la[s][i] = (unsigned)(uintptr_t)(&As[s][row * LDS_STRIDE + c]);
            lb[s][i] = (unsigned)(uintptr_t)(&Bs[s][row * LDS_STRIDE + c]);
        }
    }

#define ISSUE_TILE(BUF, KOFF)                                           \
    do {                                                                \
        _Pragma("unroll")                                               \
        for (int i = 0; i < 2; ++i) {                                   \
            async_ld_b128(la[BUF][i], pA[i] + (KOFF));                  \
            async_ld_b128(lb[BUF][i], pB[i] + (KOFF));                  \
        }                                                               \
    } while (0)

#define COMPUTE_TILE(BUF)                                               \
    do {                                                                \
        v16bf afr[4], bfr[2];                                           \
        _Pragma("unroll")                                               \
        for (int i = 0; i < 4; ++i)                                     \
            afr[i] = frag_ld(&As[BUF][(wm * 64 + i * 16) * LDS_STRIDE]);\
        _Pragma("unroll")                                               \
        for (int j = 0; j < 2; ++j)                                     \
            bfr[j] = frag_ld(&Bs[BUF][(wn * 32 + j * 16) * LDS_STRIDE]);\
        _Pragma("unroll")                                               \
        for (int i = 0; i < 4; ++i)                                     \
            _Pragma("unroll")                                           \
            for (int j = 0; j < 2; ++j)                                 \
                acc[i][j] = __builtin_amdgcn_wmma_f32_16x16x32_bf16(    \
                    false, afr[i], false, bfr[j], (short)0, acc[i][j],  \
                    false, false);                                      \
    } while (0)

    // prologue: stage K-tile 0 into buffer 0
    ISSUE_TILE(0, 0);

    // K is a multiple of 64: two 32-wide K-tiles per iteration, static buffers
    for (int k0 = 0; k0 < K; k0 += 64) {
        // even tile (buffer 0); overlap DMA of k0+32 into buffer 1
        ISSUE_TILE(1, k0 + 32);
        WAIT_ASYNC_LE4();
        __syncthreads();
        COMPUTE_TILE(0);
        __syncthreads();

        // odd tile (buffer 1); overlap DMA of k0+64 into buffer 0
        if (k0 + 64 < K) {
            ISSUE_TILE(0, k0 + 64);
            WAIT_ASYNC_LE4();
        } else {
            WAIT_ASYNC_0();
        }
        __syncthreads();
        COMPUTE_TILE(1);
        __syncthreads();
    }
#undef ISSUE_TILE
#undef COMPUTE_TILE

    // epilogue: C/D layout -> reg v: M = v + 8*(lane>>4), N = lane&15
    const int lane = tid & 31;
    const int r = lane & 15, h = lane >> 4;
#pragma unroll
    for (int i = 0; i < 4; ++i) {
#pragma unroll
        for (int j = 0; j < 2; ++j) {
#pragma unroll
            for (int v = 0; v < 8; ++v) {
                const int gm = m0 + wm * 64 + i * 16 + h * 8 + v;
                const int gn = n0 + wn * 32 + j * 16 + r;
                if (gn >= N) continue;
                float val = acc[i][j][v];
                if (MODE == 0) {
                    if (gn < D_INNER) out0[(size_t)gm * D_INNER + gn] = val;
                    else              out1[(size_t)gm * D_INNER + (gn - D_INNER)] = val;
                } else if (MODE == 1) {
                    out0[(size_t)gm * XDBL_W + gn] = val;
                    if (gn < DT_RANK) outbf[(size_t)gm * DTK + gn] = (bf16)val;
                } else if (MODE == 2) {
                    val += bias[gn];
                    val = (val > 20.f) ? val : log1pf(__expf(val));
                    out0[(size_t)gm * D_INNER + gn] = val;
                } else {
                    val += resid[(size_t)gm * D_MODEL + gn];
                    out0[(size_t)gm * D_MODEL + gn] = val;
                }
            }
        }
    }
}

// ---------------- depthwise causal conv(4) + bias + SiLU ----------------
__global__ __launch_bounds__(256) void conv_silu_kernel(
    const float* __restrict__ X, const float* __restrict__ w,
    const float* __restrict__ b, float* __restrict__ xc, bf16* __restrict__ xcbf)
{
    const size_t gid = (size_t)blockIdx.x * 256 + threadIdx.x;
    if (gid >= (size_t)MROWS * D_INNER) return;
    const int e = (int)(gid % D_INNER);
    const size_t bl = gid / D_INNER;
    const int l = (int)(bl % SEQLEN);
    float acc = b[e];
#pragma unroll
    for (int j = 0; j < D_CONV; ++j) {
        const int dl = l - (D_CONV - 1) + j;
        if (dl >= 0)
            acc += w[e * D_CONV + j] *
                   X[(bl - (size_t)(D_CONV - 1 - j)) * D_INNER + e];
    }
    const float s = acc / (1.f + __expf(-acc));
    xc[gid]   = s;
    xcbf[gid] = (bf16)s;
}

// ---------------- selective scan: 1 lane per (b, e), 16 states in VGPRs ----------------
__global__ __launch_bounds__(256) void scan_kernel(
    const float* __restrict__ dt, const float* __restrict__ xc,
    const float* __restrict__ z,  const float* __restrict__ xdbl,
    const float* __restrict__ A,  const float* __restrict__ Dskip,
    bf16* __restrict__ ybf)
{
    const int gid = blockIdx.x * 256 + threadIdx.x;   // BATCH * D_INNER lanes
    const int b = gid / D_INNER;
    const int e = gid % D_INNER;
    float a[D_STATE], h[D_STATE];
#pragma unroll
    for (int s = 0; s < D_STATE; ++s) { a[s] = A[e * D_STATE + s]; h[s] = 0.f; }
    const float dsk = Dskip[e];
    const size_t row0 = (size_t)b * SEQLEN;
    for (int l = 0; l < SEQLEN; ++l) {
        const size_t t = row0 + l;
        const float dtv = dt[t * D_INNER + e];
        const float xv  = xc[t * D_INNER + e];
        const float zv  = z [t * D_INNER + e];
        const float* pB = xdbl + t * XDBL_W + DT_RANK;
        const float* pC = pB + D_STATE;
        const float dx = dtv * xv;
        float y = 0.f;
#pragma unroll
        for (int s = 0; s < D_STATE; ++s) {
            h[s] = __expf(dtv * a[s]) * h[s] + dx * pB[s];
            y += h[s] * pC[s];
        }
        y += dsk * xv;
        y *= zv / (1.f + __expf(-zv));       // silu(z) gate
        ybf[t * D_INNER + e] = (bf16)y;
    }
}

// ---------------- LayerNorm: one wave32 per row of 768 ----------------
__global__ __launch_bounds__(256) void ln_kernel(
    const float* __restrict__ res, const float* __restrict__ gamma,
    const float* __restrict__ beta, float* __restrict__ out)
{
    const int row  = blockIdx.x * 8 + (threadIdx.x >> 5);
    const int lane = threadIdx.x & 31;
    const float* p = res + (size_t)row * D_MODEL;
    float s = 0.f, s2 = 0.f;
    for (int i = lane; i < D_MODEL; i += 32) { float v = p[i]; s += v; s2 += v * v; }
#pragma unroll
    for (int off = 16; off > 0; off >>= 1) {
        s  += __shfl_xor(s,  off, 32);
        s2 += __shfl_xor(s2, off, 32);
    }
    const float mu  = s  * (1.f / D_MODEL);
    const float var = s2 * (1.f / D_MODEL) - mu * mu;
    const float inv = rsqrtf(var + 1e-12f);
    for (int i = lane; i < D_MODEL; i += 32)
        out[(size_t)row * D_MODEL + i] = (p[i] - mu) * inv * gamma[i] + beta[i];
}

// ---------------- launch ----------------
extern "C" void kernel_launch(void* const* d_in, const int* in_sizes, int n_in,
                              void* d_out, int out_size, void* d_ws, size_t ws_size,
                              hipStream_t stream)
{
    (void)in_sizes; (void)n_in; (void)out_size; (void)ws_size;
    const float* in_x    = (const float*)d_in[0];
    const float* w_in    = (const float*)d_in[1];
    const float* w_conv  = (const float*)d_in[2];
    const float* b_conv  = (const float*)d_in[3];
    const float* w_xproj = (const float*)d_in[4];
    const float* w_dt    = (const float*)d_in[5];
    const float* b_dt    = (const float*)d_in[6];
    const float* A_log   = (const float*)d_in[7];
    const float* Dskip   = (const float*)d_in[8];
    const float* w_out   = (const float*)d_in[9];
    const float* gamma   = (const float*)d_in[10];
    const float* beta    = (const float*)d_in[11];
    float* outp = (float*)d_out;

    uint8_t* ws = (uint8_t*)d_ws;
    size_t off = 0;
    auto take = [&](size_t bytes) -> void* {
        void* p = ws + off; off += (bytes + 255) & ~(size_t)255; return p;
    };
    bf16*  in_bf = (bf16*) take((size_t)MROWS * D_MODEL * 2);
    bf16*  wi_bf = (bf16*) take((size_t)2 * D_INNER * D_MODEL * 2);
    bf16*  wx_bf = (bf16*) take((size_t)XDBL_W * D_INNER * 2);
    bf16*  wd_bf = (bf16*) take((size_t)D_INNER * DTK * 2);
    bf16*  wo_bf = (bf16*) take((size_t)D_MODEL * D_INNER * 2);
    float* Aneg  = (float*) take((size_t)D_INNER * D_STATE * 4);
    float* Xbuf  = (float*) take((size_t)MROWS * D_INNER * 4);  // reused for dt
    float* Zbuf  = (float*) take((size_t)MROWS * D_INNER * 4);
    float* XC    = (float*) take((size_t)MROWS * D_INNER * 4);  // reused for res
    bf16*  XCbf  = (bf16*) take((size_t)MROWS * D_INNER * 2);
    float* XDBL  = (float*) take((size_t)MROWS * XDBL_W * 4);
    bf16*  DTIN  = (bf16*) take((size_t)MROWS * DTK * 2);
    bf16*  YBF   = (bf16*) take((size_t)MROWS * D_INNER * 2);
    float* DT  = Xbuf;   // x (pre-conv) dead after conv
    float* RES = XC;     // xconv dead after scan

    auto nb = [](size_t n) { return (unsigned)((n + 255) / 256); };

    // prep / conversions
    cvt_f32_bf16_kernel<<<nb((size_t)MROWS * D_MODEL), 256, 0, stream>>>(in_x, in_bf, MROWS * D_MODEL);
    cvt_f32_bf16_kernel<<<nb((size_t)2 * D_INNER * D_MODEL), 256, 0, stream>>>(w_in, wi_bf, 2 * D_INNER * D_MODEL);
    cvt_f32_bf16_kernel<<<nb((size_t)XDBL_W * D_INNER), 256, 0, stream>>>(w_xproj, wx_bf, XDBL_W * D_INNER);
    cvt_f32_bf16_kernel<<<nb((size_t)D_MODEL * D_INNER), 256, 0, stream>>>(w_out, wo_bf, D_MODEL * D_INNER);
    pad_dtw_kernel<<<nb((size_t)D_INNER * DTK), 256, 0, stream>>>(w_dt, wd_bf);
    prep_A_kernel<<<nb((size_t)D_INNER * D_STATE), 256, 0, stream>>>(A_log, Aneg, D_INNER * D_STATE);
    zero_bf16_kernel<<<nb((size_t)MROWS * DTK), 256, 0, stream>>>(DTIN, MROWS * DTK);

    // 1) xz = input @ in_proj_w^T  (split into x / z)
    dim3 g0(2 * D_INNER / 128, MROWS / 128);
    gemm_bf16_wmma<0><<<g0, 256, 0, stream>>>(in_bf, wi_bf, 2 * D_INNER, D_MODEL,
                                              Xbuf, Zbuf, nullptr, nullptr, nullptr);
    // 2) causal depthwise conv + bias + SiLU
    conv_silu_kernel<<<nb((size_t)MROWS * D_INNER), 256, 0, stream>>>(Xbuf, w_conv, b_conv, XC, XCbf);
    // 3) x_dbl = xconv @ x_proj_w^T  (+ bf16 dt_in side output)
    dim3 g1(1, MROWS / 128);
    gemm_bf16_wmma<1><<<g1, 256, 0, stream>>>(XCbf, wx_bf, XDBL_W, D_INNER,
                                              XDBL, nullptr, nullptr, nullptr, DTIN);
    // 4) dt = softplus(dt_in @ dt_proj_w^T + dt_bias)
    dim3 g2(D_INNER / 128, MROWS / 128);
    gemm_bf16_wmma<2><<<g2, 256, 0, stream>>>(DTIN, wd_bf, D_INNER, DTK,
                                              DT, nullptr, b_dt, nullptr, nullptr);
    // 5) selective scan + D-skip + silu(z) gate -> bf16 y
    scan_kernel<<<(BATCH * D_INNER) / 256, 256, 0, stream>>>(DT, XC, Zbuf, XDBL, Aneg, Dskip, YBF);
    // 6) hidden = y @ out_proj_w^T + residual
    dim3 g3(D_MODEL / 128, MROWS / 128);
    gemm_bf16_wmma<3><<<g3, 256, 0, stream>>>(YBF, wo_bf, D_MODEL, D_INNER,
                                              RES, nullptr, nullptr, in_x, nullptr);
    // 7) LayerNorm
    ln_kernel<<<MROWS / 8, 256, 0, stream>>>(RES, gamma, beta, outp);
}
